// GHRNN_12506944766401
// MI455X (gfx1250) — matrix operand order repaired and checked by reading, not compiled, not run on hardware
//
#include <hip/hip_runtime.h>
#include <hip/hip_bf16.h>

typedef __attribute__((ext_vector_type(16))) __bf16 v16bf;
typedef __attribute__((ext_vector_type(8)))  float  v8f;

union FragBF {
    v16bf v;
    uint4 q[2];
    unsigned short u[16];
};

__device__ __forceinline__ unsigned short f2bf(float f) {
    unsigned int u = __float_as_uint(f);
    unsigned int r = u + 0x7FFFu + ((u >> 16) & 1u);   // round-to-nearest-even
    return (unsigned short)(r >> 16);
}
__device__ __forceinline__ float bf2f(unsigned short b) {
    return __uint_as_float(((unsigned int)b) << 16);
}

__device__ __forceinline__ v8f wmma_bf16(v16bf a, v16bf b, v8f c) {
    // D = A(16x32 bf16) * B(32x16 bf16) + C(16x16 f32)
    return __builtin_amdgcn_wmma_f32_16x16x32_bf16(
        /*neg_a=*/false, a, /*neg_b=*/false, b,
        /*c_mod=*/(short)0, c, /*reuse_a=*/false, /*reuse_b=*/false);
}

// ---- fragment loaders (CDNA5 ISA 7.12.2 layouts) ---------------------------

// A: 16x32 bf16 tile of row-major A[M,K].  lanes 0-15: rows, K {k0..k0+7, k0+16..k0+23};
// lanes 16-31: same rows, K {k0+8..k0+15, k0+24..k0+31}.
__device__ __forceinline__ v16bf load_a_bf16(const unsigned short* A, int lda,
                                             int m0, int k0, int lane) {
    int row = m0 + (lane & 15);
    int kb  = k0 + ((lane & 16) ? 8 : 0);
    const unsigned short* p = A + (size_t)row * lda + kb;
    FragBF f;
    f.q[0] = *(const uint4*)(p);        // K = kb .. kb+7
    f.q[1] = *(const uint4*)(p + 16);   // K = kb+16 .. kb+23
    return f.v;
}

// A fragment from an LDS-resident 16x32 tile (row-major, lda = 32).
__device__ __forceinline__ v16bf load_a_lds(const unsigned short* As, int lane) {
    int row = lane & 15;
    int kb  = (lane & 16) ? 8 : 0;
    const unsigned short* p = As + row * 32 + kb;
    FragBF f;
    f.q[0] = *(const uint4*)(p);
    f.q[1] = *(const uint4*)(p + 16);
    return f.v;
}

// A fragment from f32 source, converted to bf16 in-register.
__device__ __forceinline__ v16bf load_a_f32(const float* A, int lda,
                                            int m0, int k0, int lane) {
    int row = m0 + (lane & 15);
    int kb  = k0 + ((lane & 16) ? 8 : 0);
    const float* p = A + (size_t)row * lda + kb;
    float4 a0 = *(const float4*)(p);
    float4 a1 = *(const float4*)(p + 4);
    float4 a2 = *(const float4*)(p + 16);
    float4 a3 = *(const float4*)(p + 20);
    FragBF f;
    f.u[0]  = f2bf(a0.x); f.u[1]  = f2bf(a0.y); f.u[2]  = f2bf(a0.z); f.u[3]  = f2bf(a0.w);
    f.u[4]  = f2bf(a1.x); f.u[5]  = f2bf(a1.y); f.u[6]  = f2bf(a1.z); f.u[7]  = f2bf(a1.w);
    f.u[8]  = f2bf(a2.x); f.u[9]  = f2bf(a2.y); f.u[10] = f2bf(a2.z); f.u[11] = f2bf(a2.w);
    f.u[12] = f2bf(a3.x); f.u[13] = f2bf(a3.y); f.u[14] = f2bf(a3.z); f.u[15] = f2bf(a3.w);
    return f.v;
}

// B: 32x16 bf16 tile where B[k][n] = W[n0+n][k0+k], W row-major [N,K].
// lane = column n: lanes 0-15 K {k0..k0+7, k0+16..k0+23}; lanes 16-31 K {k0+8..k0+15, k0+24..k0+31}.
__device__ __forceinline__ v16bf load_b_bf16(const unsigned short* W, int ldw,
                                             int n0, int k0, int lane) {
    int nrow = n0 + (lane & 15);
    int kb   = k0 + ((lane & 16) ? 8 : 0);
    const unsigned short* p = W + (size_t)nrow * ldw + kb;
    FragBF f;
    f.q[0] = *(const uint4*)(p);
    f.q[1] = *(const uint4*)(p + 16);
    return f.v;
}

// ---- kernels ---------------------------------------------------------------

__global__ __launch_bounds__(256)
void cvt_bf16_kernel(const float* __restrict__ s, unsigned short* __restrict__ d, int n) {
    int i = blockIdx.x * blockDim.x + threadIdx.x;
    if (i < n) d[i] = f2bf(s[i]);
}

// C[M,N] f32 = A_bf16[M,K] @ W_bf16[N,K]^T + bias.
// One 16x64 output strip per wave: A fragment reused across 4 accumulators.
__global__ __launch_bounds__(256)
void gemm_h2out_kernel(const unsigned short* __restrict__ A,
                       const unsigned short* __restrict__ W,
                       const float* __restrict__ bias,
                       float* __restrict__ C,
                       int M, int N, int K) {
    int lane = threadIdx.x & 31;
    int tile = blockIdx.x * 8 + (threadIdx.x >> 5);
    int ntn  = N >> 6;                       // strips of 64 columns
    int m0   = (tile / ntn) << 4;
    int n0   = (tile % ntn) << 6;
    if (m0 >= M) return;
    v8f acc[4] = {};
#pragma unroll 2
    for (int k = 0; k < K; k += 32) {
        v16bf a = load_a_bf16(A, K, m0, k, lane);
#pragma unroll
        for (int j = 0; j < 4; j++)
            acc[j] = wmma_bf16(a, load_b_bf16(W, K, n0 + (j << 4), k, lane), acc[j]);
    }
    int rbase = m0 + ((lane & 16) ? 8 : 0);
#pragma unroll
    for (int j = 0; j < 4; j++) {
        int col  = n0 + (j << 4) + (lane & 15);
        float bv = bias[col];
#pragma unroll
        for (int r = 0; r < 8; r++)
            C[(size_t)(rbase + r) * N + col] = acc[j][r] + bv;
    }
}

// Xe[M,N] bf16 = X_f32[M,K] @ W_bf16[N,K]^T + bias   (input embeddings)
__global__ __launch_bounds__(256)
void gemm_x2e_kernel(const float* __restrict__ X,
                     const unsigned short* __restrict__ W,
                     const float* __restrict__ bias,
                     unsigned short* __restrict__ Xe,
                     int M, int N, int K) {
    int lane = threadIdx.x & 31;
    int tile = blockIdx.x * 8 + (threadIdx.x >> 5);
    int ntn  = N >> 6;
    int m0   = (tile / ntn) << 4;
    int n0   = (tile % ntn) << 6;
    if (m0 >= M) return;
    v8f acc[4] = {};
#pragma unroll 2
    for (int k = 0; k < K; k += 32) {
        v16bf a = load_a_f32(X, K, m0, k, lane);
#pragma unroll
        for (int j = 0; j < 4; j++)
            acc[j] = wmma_bf16(a, load_b_bf16(W, K, n0 + (j << 4), k, lane), acc[j]);
    }
    int rbase = m0 + ((lane & 16) ? 8 : 0);
#pragma unroll
    for (int j = 0; j < 4; j++) {
        int col  = n0 + (j << 4) + (lane & 15);
        float bv = bias[col];
#pragma unroll
        for (int r = 0; r < 8; r++)
            Xe[(size_t)(rbase + r) * N + col] = f2bf(acc[j][r] + bv);
    }
}

// Fused GRU step.  Block = 8 waves sharing one 16-token tile; the A tile
// (16x32 bf16, 1KB) is double-buffered in LDS and shared by all waves, each
// wave owns 16 hidden columns and accumulates the r/z/n gate pre-activations
// (3 WMMAs per A fragment).
__global__ __launch_bounds__(256)
void gru_step_kernel(const unsigned short* __restrict__ hOld,  // [T,1024] bf16
                     const unsigned short* __restrict__ xE,    // [T,512]  bf16
                     const unsigned short* __restrict__ Wih,   // [3072,512]
                     const unsigned short* __restrict__ Whh,   // [3072,1024]
                     const float* __restrict__ bih,            // [3072]
                     const float* __restrict__ bhh,            // [3072]
                     unsigned short* __restrict__ hNew) {      // [T,1024] bf16
    const int Hd = 1024, Ed = 512;
    __shared__ alignas(16) unsigned short As[2][16 * 32];

    int lane = threadIdx.x & 31;
    int wv   = threadIdx.x >> 5;
    int m0   = blockIdx.x << 4;                  // token tile
    int n0   = (blockIdx.y << 7) + (wv << 4);    // hidden-column tile

    // stage a 16x32 bf16 A tile into LDS buffer `buf` (64 threads x 16B)
    auto stage = [&](const unsigned short* src, int lds_, int k, int buf) {
        int t = threadIdx.x;
        if (t < 64) {
            int row = t >> 2, ch = (t & 3) << 3;
            *(uint4*)&As[buf][row * 32 + ch] =
                *(const uint4*)(src + (size_t)(m0 + row) * lds_ + k + ch);
        }
    };

    v8f ar = {}, az = {}, ain = {}, ahn = {};

    // ---- gi = x @ W_ih^T  (gate rows: r=n0, z=H+n0, n=2H+n0) ----
    stage(xE, Ed, 0, 0);
    for (int k = 0; k < Ed; k += 32) {
        int buf = (k >> 5) & 1;
        __syncthreads();                          // As[buf] ready; As[buf^1] free
        if (k + 32 < Ed) stage(xE, Ed, k + 32, buf ^ 1);
        v16bf a = load_a_lds(&As[buf][0], lane);
        ar  = wmma_bf16(a, load_b_bf16(Wih, Ed, n0,          k, lane), ar);
        az  = wmma_bf16(a, load_b_bf16(Wih, Ed, Hd + n0,     k, lane), az);
        ain = wmma_bf16(a, load_b_bf16(Wih, Ed, 2 * Hd + n0, k, lane), ain);
    }
    // ---- gh = h @ W_hh^T ----
    stage(hOld, Hd, 0, 0);                        // last loop read buf=1; safe
    for (int k = 0; k < Hd; k += 32) {
        int buf = (k >> 5) & 1;
        __syncthreads();
        if (k + 32 < Hd) stage(hOld, Hd, k + 32, buf ^ 1);
        v16bf a = load_a_lds(&As[buf][0], lane);
        ar  = wmma_bf16(a, load_b_bf16(Whh, Hd, n0,          k, lane), ar);
        az  = wmma_bf16(a, load_b_bf16(Whh, Hd, Hd + n0,     k, lane), az);
        ahn = wmma_bf16(a, load_b_bf16(Whh, Hd, 2 * Hd + n0, k, lane), ahn);
    }

    int col   = n0 + (lane & 15);
    int rbase = m0 + ((lane & 16) ? 8 : 0);
    float br  = bih[col] + bhh[col];
    float bz  = bih[Hd + col] + bhh[Hd + col];
    float bin = bih[2 * Hd + col];
    float bhn = bhh[2 * Hd + col];
#pragma unroll
    for (int r = 0; r < 8; r++) {
        int row  = rbase + r;
        float rg = 1.f / (1.f + __expf(-(ar[r] + br)));
        float zg = 1.f / (1.f + __expf(-(az[r] + bz)));
        float ng = tanhf(ain[r] + bin + rg * (ahn[r] + bhn));
        float ho = bf2f(hOld[(size_t)row * Hd + col]);
        hNew[(size_t)row * Hd + col] = f2bf((1.f - zg) * ng + zg * ho);
    }
}

// ---- host ------------------------------------------------------------------

extern "C" void kernel_launch(void* const* d_in, const int* in_sizes, int n_in,
                              void* d_out, int out_size, void* d_ws, size_t ws_size,
                              hipStream_t stream) {
    (void)in_sizes; (void)n_in; (void)out_size; (void)ws_size;
    constexpr int T  = 64 * 256;   // B*S tokens
    constexpr int Hd = 1024, Ed = 512;

    const float* hidden = (const float*)d_in[0];
    const float* x_t1 = (const float*)d_in[1];
    const float* x_t2 = (const float*)d_in[2];
    const float* x_v1 = (const float*)d_in[3];
    const float* x_e  = (const float*)d_in[4];
    const float* Wt1 = (const float*)d_in[5];  const float* bt1 = (const float*)d_in[6];
    const float* Wt2 = (const float*)d_in[7];  const float* bt2 = (const float*)d_in[8];
    const float* Wv1 = (const float*)d_in[9];  const float* bv1 = (const float*)d_in[10];
    const float* We  = (const float*)d_in[11]; const float* be  = (const float*)d_in[12];
    const float* W_ih = (const float*)d_in[13];
    const float* W_hh = (const float*)d_in[14];
    const float* b_ih = (const float*)d_in[15];
    const float* b_hh = (const float*)d_in[16];
    const float* O1w = (const float*)d_in[17]; const float* O1b = (const float*)d_in[18];
    const float* O2w = (const float*)d_in[19]; const float* O2b = (const float*)d_in[20];
    const float* O3w = (const float*)d_in[21]; const float* O3b = (const float*)d_in[22];
    const float* O4w = (const float*)d_in[23]; const float* O4b = (const float*)d_in[24];
    const float* O5w = (const float*)d_in[25]; const float* O5b = (const float*)d_in[26];
    float* out = (float*)d_out;

    char* p = (char*)d_ws;
    auto carve = [&](size_t elems) -> unsigned short* {
        unsigned short* r = (unsigned short*)p;
        p += (elems * 2 + 255) & ~(size_t)255;
        return r;
    };
    unsigned short* hA  = carve((size_t)T * Hd);
    unsigned short* hB  = carve((size_t)T * Hd);
    unsigned short* xe  = carve((size_t)T * Ed);
    unsigned short* wt1 = carve(512 * 64);
    unsigned short* wt2 = carve(512 * 64);
    unsigned short* wv1 = carve(512 * 256);
    unsigned short* web = carve(512 * 256);
    unsigned short* wih = carve(3072 * 512);
    unsigned short* whh = carve(3072 * 1024);
    unsigned short* o1  = carve(64 * 1024);
    unsigned short* o2  = carve(64 * 1024);
    unsigned short* o3  = carve(256 * 1024);
    unsigned short* o4  = carve(256 * 1024);
    unsigned short* o5  = carve(256 * 1024);

    auto cvt = [&](const float* s, unsigned short* d, int n) {
        cvt_bf16_kernel<<<(n + 255) / 256, 256, 0, stream>>>(s, d, n);
    };
    cvt(hidden, hA, T * Hd);
    cvt(Wt1, wt1, 512 * 64);   cvt(Wt2, wt2, 512 * 64);
    cvt(Wv1, wv1, 512 * 256);  cvt(We,  web, 512 * 256);
    cvt(W_ih, wih, 3072 * 512); cvt(W_hh, whh, 3072 * 1024);
    cvt(O1w, o1, 64 * 1024);   cvt(O2w, o2, 64 * 1024);
    cvt(O3w, o3, 256 * 1024);  cvt(O4w, o4, 256 * 1024);
    cvt(O5w, o5, 256 * 1024);

    auto head = [&](const unsigned short* h, const unsigned short* Ow,
                    const float* Ob, float* Cout, int N) {
        int tiles = (T / 16) * (N / 64);
        gemm_h2out_kernel<<<(tiles + 7) / 8, 256, 0, stream>>>(h, Ow, Ob, Cout, T, N, Hd);
    };
    auto embed = [&](const float* x, const unsigned short* Wb, const float* bb, int Kin) {
        int tiles = (T / 16) * (Ed / 64);
        gemm_x2e_kernel<<<(tiles + 7) / 8, 256, 0, stream>>>(x, Wb, bb, xe, T, Ed, Kin);
    };
    auto gru = [&](const unsigned short* hO, unsigned short* hN) {
        gru_step_kernel<<<dim3(T / 16, Hd / 128), 256, 0, stream>>>(
            hO, xe, wih, whh, b_ih, b_hh, hN);
    };

    size_t oT1 = 0;
    size_t oT2 = (size_t)T * 64;
    size_t oV1 = oT2 + (size_t)T * 64;
    size_t oE  = oV1 + (size_t)T * 256;
    size_t oV2 = oE  + (size_t)T * 256;

    head(hA, o1, O1b, out + oT1, 64);      // all_t1 = h0 @ Ot1
    embed(x_t1, wt1, bt1, 64);
    gru(hA, hB);                            // h1
    head(hB, o2, O2b, out + oT2, 64);      // all_t2
    embed(x_t2, wt2, bt2, 64);
    gru(hB, hA);                            // h2
    head(hA, o3, O3b, out + oV1, 256);     // all_v1
    embed(x_v1, wv1, bv1, 256);
    gru(hA, hB);                            // h3
    head(hB, o4, O4b, out + oE, 256);      // all_e
    embed(x_e, web, be, 256);
    gru(hB, hA);                            // h4
    head(hA, o5, O5b, out + oV2, 256);     // all_v2
}
